// SelfAttentionBlock_79276506349922
// MI455X (gfx1250) — matrix operand order repaired and checked by reading, not compiled, not run on hardware
//
#include <hip/hip_runtime.h>

#define B_DIM 8
#define T_DIM 2048
#define C_DIM 1024
#define H_DIM 128

// Toolchain-dependent TDM builtin arity (probe-verified):
//   ROCm 7.2 / AMD clang-22 : 5 args (g0 v4u, g1 v8i, g2 v4i, g3 v4i, cpol)
//   amdgpu-toolchain clang-23: 6 args (g0 v4u, g1 v8i, g2 v4i, g3 v4i, v8i, cpol)
#if defined(__clang_major__) && (__clang_major__ >= 23)
#define TDM_6ARG 1
#else
#define TDM_6ARG 0
#endif

typedef __attribute__((ext_vector_type(16))) __bf16 v16bf;
typedef __attribute__((ext_vector_type(8)))  float  v8f;
typedef unsigned int v4u __attribute__((ext_vector_type(4)));
typedef int          v4i __attribute__((ext_vector_type(4)));
typedef int          v8i __attribute__((ext_vector_type(8)));

union FragBF {
    v16bf v;
    unsigned u[8];
    uint4 q[2];
};

__device__ __forceinline__ unsigned short bf16_bits(float f) {
    return __builtin_bit_cast(unsigned short, static_cast<__bf16>(f));
}
__device__ __forceinline__ unsigned pack_bf16(float lo, float hi) {
    return (unsigned)bf16_bits(lo) | ((unsigned)bf16_bits(hi) << 16);
}
__device__ __forceinline__ v8f vzero8() {
    v8f z;
#pragma unroll
    for (int i = 0; i < 8; ++i) z[i] = 0.0f;
    return z;
}
// A-fragment (16x32 bf16): 16 elems per lane at row*stride + kbase, split {0..7} and {16..23}
__device__ __forceinline__ FragBF load_frag_a16(const unsigned short* p) {
    FragBF f;
    f.q[0] = *(const uint4*)(p);
    f.q[1] = *(const uint4*)(p + 16);
    return f;
}
// B-fragment (32x16 bf16): 16 K elems per lane at kbase, split {0..7} and {16..23}
__device__ __forceinline__ FragBF load_frag_b16(const unsigned short* p) {
    FragBF f;
    f.q[0] = *(const uint4*)(p);
    f.q[1] = *(const uint4*)(p + 16);
    return f;
}
__device__ __forceinline__ v8f wmma_bf16(const FragBF& a, const FragBF& b, v8f c) {
    return __builtin_amdgcn_wmma_f32_16x16x32_bf16(false, a.v, false, b.v, (short)0, c,
                                                   false, false);
}

// Low 32 bits of a generic pointer to LDS = workgroup-relative LDS byte address
// (flat-LDS aperture keeps the local offset in addr[31:0]).
__device__ __forceinline__ unsigned lds_offset(const void* p) {
    return (unsigned)(unsigned long long)p;
}
// TDM 2D tile load: tile_h rows x tile_w bf16 elements, row stride = stride elems.
// Descriptor per CDNA5 ISA (groups 0/1; 2D so groups 2/3 zero).
__device__ __forceinline__ void tdm_load_2d_bf16(unsigned lds_addr, const void* gaddr,
                                                 unsigned tile_w, unsigned tile_h,
                                                 unsigned stride) {
    unsigned long long ga = (unsigned long long)gaddr;
    v4u g0;
    g0.x = 1u;                                      // count=1, user descriptor
    g0.y = lds_addr;                                // lds_addr (bytes)
    g0.z = (unsigned)(ga & 0xFFFFFFFFull);          // global_addr[31:0]
    g0.w = (unsigned)((ga >> 32) & 0x01FFFFFFull)   // global_addr[56:32]
           | (2u << 30);                            // type = 2 ("image")
    v8i g1;
    g1[0] = (int)(1u << 16);                        // wg_mask=0, data_size=1 (2 bytes)
    g1[1] = (int)((tile_w & 0xFFFFu) << 16);        // tensor_dim0[15:0] @ bits 63:48
    g1[2] = (int)(((tile_w >> 16) & 0xFFFFu)        // tensor_dim0[31:16]
           | ((tile_h & 0xFFFFu) << 16));           // tensor_dim1[15:0]
    g1[3] = (int)(((tile_h >> 16) & 0xFFFFu)        // tensor_dim1[31:16]
           | ((tile_w & 0xFFFFu) << 16));           // tile_dim0
    g1[4] = (int)(tile_h & 0xFFFFu);                // tile_dim1 (tile_dim2 = 0)
    g1[5] = (int)stride;                            // tensor_dim0_stride[31:0]
    g1[6] = 0;                                      // stride[47:32], dim1_stride lo = 0
    g1[7] = 0;
    v4i zz = {0, 0, 0, 0};
#if TDM_6ARG
    v8i z8 = {0, 0, 0, 0, 0, 0, 0, 0};
    __builtin_amdgcn_tensor_load_to_lds(g0, g1, zz, zz, z8, 0);
#else
    __builtin_amdgcn_tensor_load_to_lds(g0, g1, zz, zz, 0);
#endif
}

// ---------------------------------------------------------------------------
// Kernel 1: QKV projection.  x[B*T,C] f32  x  W[C,H] f32  ->  bf16 [B*T,H].
// 4 waves per block, 16 rows per wave, 64 rows per block.
// W chunk (32 x 128) staged in LDS transposed (f32 -> bf16 on the fly).
// ---------------------------------------------------------------------------
__global__ __launch_bounds__(128) void qkv_proj_kernel(
    const float* __restrict__ x,
    const float* __restrict__ Wq, const float* __restrict__ Wk, const float* __restrict__ Wv,
    unsigned short* __restrict__ qb, unsigned short* __restrict__ kb,
    unsigned short* __restrict__ vb)
{
    __shared__ unsigned short Wt[H_DIM][32];   // Wt[h][k] = W[k0+k][h]

    const int tid   = threadIdx.x;
    const int wave  = tid >> 5;
    const int lane  = tid & 31;
    const int row0  = blockIdx.x * 64 + wave * 16;     // row in [0, B*T)

    const int lhalf = (lane < 16) ? 0 : 1;
    const int am    = lane & 15;           // A-matrix row (M)
    const int akb   = lhalf ? 8 : 0;       // A K-base
    const int bn    = lane & 15;           // B-matrix column (N)
    const int bkb   = lhalf ? 8 : 0;       // B K-base

    const float* Ws[3]    = {Wq, Wk, Wv};
    unsigned short* Os[3] = {qb, kb, vb};

    for (int w = 0; w < 3; ++w) {
        const float* W = Ws[w];
        v8f acc[8];
#pragma unroll
        for (int nt = 0; nt < 8; ++nt) acc[nt] = vzero8();

        for (int k0 = 0; k0 < C_DIM; k0 += 32) {
            __syncthreads();   // protect previous chunk's readers
            // stage W[k0:k0+32, 0:128] -> Wt transposed, converting to bf16
#pragma unroll
            for (int i = 0; i < 8; ++i) {
                int chunk = tid + 128 * i;        // 0..1023
                int kr    = chunk >> 5;           // 0..31
                int c4    = (chunk & 31) << 2;    // 0..124
                float4 f  = *(const float4*)(W + (size_t)(k0 + kr) * H_DIM + c4);
                Wt[c4 + 0][kr] = bf16_bits(f.x);
                Wt[c4 + 1][kr] = bf16_bits(f.y);
                Wt[c4 + 2][kr] = bf16_bits(f.z);
                Wt[c4 + 3][kr] = bf16_bits(f.w);
            }
            __syncthreads();

            // A fragment: convert x rows f32 -> bf16 in registers
            FragBF a;
            {
                const float* xr = x + (size_t)(row0 + am) * C_DIM + k0 + akb;
                float4 f0 = *(const float4*)(xr);
                float4 f1 = *(const float4*)(xr + 4);
                float4 f2 = *(const float4*)(xr + 16);
                float4 f3 = *(const float4*)(xr + 20);
                a.u[0] = pack_bf16(f0.x, f0.y); a.u[1] = pack_bf16(f0.z, f0.w);
                a.u[2] = pack_bf16(f1.x, f1.y); a.u[3] = pack_bf16(f1.z, f1.w);
                a.u[4] = pack_bf16(f2.x, f2.y); a.u[5] = pack_bf16(f2.z, f2.w);
                a.u[6] = pack_bf16(f3.x, f3.y); a.u[7] = pack_bf16(f3.z, f3.w);
            }
#pragma unroll
            for (int nt = 0; nt < 8; ++nt) {
                FragBF bf = load_frag_b16(&Wt[nt * 16 + bn][bkb]);
                acc[nt] = wmma_bf16(a, bf, acc[nt]);
            }
        }
        // store D tiles as bf16: col = bn, row = v + 8*lhalf
        unsigned short* O = Os[w];
#pragma unroll
        for (int nt = 0; nt < 8; ++nt) {
#pragma unroll
            for (int v = 0; v < 8; ++v) {
                int r = v + 8 * lhalf;
                O[(size_t)(row0 + r) * H_DIM + nt * 16 + bn] = bf16_bits(acc[nt][v]);
            }
        }
    }
}

// ---------------------------------------------------------------------------
// Kernel 2: causal flash attention over bf16 q/k/v, f32 output.
// Grid (T/64, B); 4 waves per block; each wave owns 16 query rows.
// Key blocks of 32: K staged row-major via TDM (tensor_load_to_lds),
// V staged transposed via VGPRs (TDM cannot transpose elements).
// ---------------------------------------------------------------------------
__global__ __launch_bounds__(128) void flash_attn_kernel(
    const unsigned short* __restrict__ qb,
    const unsigned short* __restrict__ kb,
    const unsigned short* __restrict__ vb,
    float* __restrict__ out)
{
    __shared__ unsigned short Kt[32][H_DIM];       // K rows (key-major, H contiguous)
    __shared__ unsigned short Vt[H_DIM][32];       // V transposed: Vt[h][key]
    __shared__ unsigned short Ps[4][16][32];       // per-wave P scratch (bf16)

    const int tid   = threadIdx.x;
    const int wave  = tid >> 5;
    const int lane  = tid & 31;
    const int b     = blockIdx.y;
    const int tile  = blockIdx.x;
    const int tq    = tile * 64 + wave * 16;       // wave's query base in [0,T)
    const size_t rowQ = (size_t)b * T_DIM + tq;

    const int lhalf = (lane < 16) ? 0 : 1;
    const int am    = lane & 15;
    const int akb   = lhalf ? 8 : 0;
    const int bn    = lane & 15;
    const int bkb   = lhalf ? 8 : 0;

    // Q strip resident in registers: 4 A-fragments over H
    FragBF qf[4];
#pragma unroll
    for (int h = 0; h < 4; ++h)
        qf[h] = load_frag_a16(qb + (rowQ + am) * H_DIM + h * 32 + akb);

    v8f o[8];
#pragma unroll
    for (int nt = 0; nt < 8; ++nt) o[nt] = vzero8();
    float mrow[8], lrow[8];
#pragma unroll
    for (int v = 0; v < 8; ++v) { mrow[v] = -1e30f; lrow[v] = 0.0f; }

    const float scale = 0.03125f;                  // C^-0.5 = 1/32
    const int   jend  = tile * 64 + 64;

    for (int j0 = 0; j0 < jend; j0 += 32) {
        __syncthreads();                           // previous block's readers done
        // K block: one wave kicks the Tensor Data Mover (32 x 128 bf16 2D tile);
        // the DMA overlaps with the VALU/DS transpose-staging of V below.
        if (wave == 0)
            tdm_load_2d_bf16(lds_offset(&Kt[0][0]),
                             kb + ((size_t)b * T_DIM + j0) * H_DIM,
                             H_DIM, 32, H_DIM);
        // V block staged transposed by all threads
#pragma unroll
        for (int i = 0; i < 4; ++i) {
            int chunk = tid + 128 * i;     // 0..511
            int r     = chunk >> 4;        // 0..31
            int c8    = (chunk & 15) << 3; // 0..120
            size_t g  = ((size_t)b * T_DIM + j0 + r) * H_DIM + c8;
            uint4 dv  = *(const uint4*)(vb + g);
            const unsigned short* s = (const unsigned short*)&dv;
#pragma unroll
            for (int e = 0; e < 8; ++e) Vt[c8 + e][r] = s[e];
        }
        if (j0 + 32 < jend) {
            size_t gn = ((size_t)b * T_DIM + j0 + 32 + (tid & 31)) * H_DIM + ((tid >> 5) << 5);
            __builtin_prefetch(kb + gn, 0, 1);
            __builtin_prefetch(vb + gn, 0, 1);
        }
        if (wave == 0) __builtin_amdgcn_s_wait_tensorcnt(0);
        __syncthreads();

        if (j0 > tq + 15) continue;        // block fully above causal diagonal for this wave

        // S = Q K^T : two 16x16 column tiles
        v8f s0 = vzero8(), s1 = vzero8();
#pragma unroll
        for (int h = 0; h < 4; ++h) {
            FragBF kf0 = load_frag_b16(&Kt[bn][h * 32 + bkb]);
            FragBF kf1 = load_frag_b16(&Kt[16 + bn][h * 32 + bkb]);
            s0 = wmma_bf16(qf[h], kf0, s0);
            s1 = wmma_bf16(qf[h], kf1, s1);
        }

        // scale + causal mask + online softmax
        float rmax[8];
#pragma unroll
        for (int v = 0; v < 8; ++v) {
            int tqr = tq + v + 8 * lhalf;
            float a0 = s0[v] * scale; if (j0 + bn      > tqr) a0 = -1e30f;
            float a1 = s1[v] * scale; if (j0 + 16 + bn > tqr) a1 = -1e30f;
            s0[v] = a0; s1[v] = a1;
            float mx = fmaxf(a0, a1);
            mx = fmaxf(mx, __shfl_xor(mx, 1, 32));
            mx = fmaxf(mx, __shfl_xor(mx, 2, 32));
            mx = fmaxf(mx, __shfl_xor(mx, 4, 32));
            mx = fmaxf(mx, __shfl_xor(mx, 8, 32));
            rmax[v] = mx;
        }
#pragma unroll
        for (int v = 0; v < 8; ++v) {
            float mnew  = fmaxf(mrow[v], rmax[v]);
            float alpha = __expf(mrow[v] - mnew);
            mrow[v] = mnew;
            float p0 = __expf(s0[v] - mnew);
            float p1 = __expf(s1[v] - mnew);
            float rs = p0 + p1;
            rs += __shfl_xor(rs, 1, 32);
            rs += __shfl_xor(rs, 2, 32);
            rs += __shfl_xor(rs, 4, 32);
            rs += __shfl_xor(rs, 8, 32);
            lrow[v] = lrow[v] * alpha + rs;
#pragma unroll
            for (int nt = 0; nt < 8; ++nt) o[nt][v] *= alpha;
            int r = v + 8 * lhalf;
            Ps[wave][r][bn]      = bf16_bits(p0);
            Ps[wave][r][16 + bn] = bf16_bits(p1);
        }
        asm volatile("" ::: "memory");     // keep DS stores before DS loads (wave-private tile)

        // O += P V
        FragBF pf = load_frag_a16(&Ps[wave][am][akb]);
#pragma unroll
        for (int nt = 0; nt < 8; ++nt) {
            FragBF vf = load_frag_b16(&Vt[nt * 16 + bn][bkb]);
            o[nt] = wmma_bf16(pf, vf, o[nt]);
        }
    }

    // normalize and store f32 output
#pragma unroll
    for (int nt = 0; nt < 8; ++nt) {
#pragma unroll
        for (int v = 0; v < 8; ++v) {
            int r = v + 8 * lhalf;
            out[(rowQ + r) * H_DIM + nt * 16 + bn] = o[nt][v] / lrow[v];
        }
    }
}

// ---------------------------------------------------------------------------
extern "C" void kernel_launch(void* const* d_in, const int* in_sizes, int n_in,
                              void* d_out, int out_size, void* d_ws, size_t ws_size,
                              hipStream_t stream) {
    (void)in_sizes; (void)n_in; (void)out_size; (void)ws_size;
    const float* x  = (const float*)d_in[0];
    const float* Wk = (const float*)d_in[1];
    const float* Wq = (const float*)d_in[2];
    const float* Wv = (const float*)d_in[3];

    const size_t NP = (size_t)B_DIM * T_DIM * H_DIM;   // 2M elems per projection
    unsigned short* qb = (unsigned short*)d_ws;
    unsigned short* kb = qb + NP;
    unsigned short* vb = kb + NP;

    qkv_proj_kernel<<<(B_DIM * T_DIM) / 64, 128, 0, stream>>>(x, Wq, Wk, Wv, qb, kb, vb);

    dim3 grid(T_DIM / 64, B_DIM, 1);
    flash_attn_kernel<<<grid, 128, 0, stream>>>(qb, kb, vb, (float*)d_out);
}
